// MambaPatchExpand_21569325760565
// MI455X (gfx1250) — compile-verified, matrix operand order checked by reading, not run on hardware
//
#include <hip/hip_runtime.h>
#include <hip/hip_bf16.h>

// ---------------------------------------------------------------------------
// VMamba SS2D forward for MI455X (gfx1250), wave32 + WMMA bf16.
// Pipeline: in_proj GEMM (WMMA) -> dwconv3x3+SiLU+4-dir scatter ->
//           x_dbl GEMM (WMMA, padded weights) -> 3-pass chunked selective scan
//           -> combine+LayerNorm+SiLU-gate -> out_proj GEMM (WMMA).
// All B-tile loads are unguarded (weights pre-padded) so EXEC stays all-ones
// around every v_wmma. Each wave keeps its A fragment in registers and sweeps
// multiple N tiles per K-step (4/3/6 accumulators).
// Workspace required: ~107 MB + 36,864 floats for padded weights.
// ---------------------------------------------------------------------------

typedef __attribute__((ext_vector_type(16))) __bf16 v16bf;
typedef __attribute__((ext_vector_type(8)))  float  v8f;

constexpr int BQ    = 4;      // batch
constexpr int HH    = 64;
constexpr int WWID  = 64;
constexpr int LSEQ  = HH * WWID;       // 4096
constexpr int DMODEL= 96;
constexpr int DI    = 192;             // d_inner
constexpr int KDIR  = 4;
constexpr int NST   = 16;              // d_state
constexpr int DTR   = 6;               // dt_rank
constexpr int CDBL  = DTR + 2 * NST;   // 38
constexpr int XLD   = 48;              // padded x_dbl row stride
constexpr int NC    = 32;              // scan chunks
constexpr int CLEN  = LSEQ / NC;       // 128

// ---------------- WMMA helpers (ISA 7.12.2 VGPR layouts) -------------------

__device__ __forceinline__ v8f wmma_bf16(v16bf a, v16bf b, v8f c) {
  return __builtin_amdgcn_wmma_f32_16x16x32_bf16(
      /*neg_a=*/false, a, /*neg_b=*/false, b,
      /*c_mod=*/(short)0, c, /*reuse_a=*/false, /*reuse_b=*/false);
}

// A tile: 16x32 bf16 from row-major fp32 [m0..m0+15, k0..k0+31], stride lda.
// lanes 0-15: row M=lane, K = k0..k0+7 (elems 0-7), k0+16..k0+23 (elems 8-15)
// lanes 16-31: row M=lane-16, K = k0+8..15 and k0+24..31.
__device__ __forceinline__ v16bf load_a_rm(const float* __restrict__ A, int lda,
                                           int m0, int k0, int lane) {
  const float* p = A + (size_t)(m0 + (lane & 15)) * lda + k0 + ((lane & 16) ? 8 : 0);
  v16bf r;
#pragma unroll
  for (int i = 0; i < 8; ++i) r[i] = (__bf16)p[i];
#pragma unroll
  for (int i = 0; i < 8; ++i) r[8 + i] = (__bf16)p[16 + i];
  return r;
}

// B tile: 32x16 bf16 where B[k][n] = W[n][k] (we compute X @ W^T).
// Mirrored layout: lane indexes N, same K packing as A. Unguarded.
__device__ __forceinline__ v16bf load_bT_rm(const float* __restrict__ W, int ldw,
                                            int n0, int k0, int lane) {
  const float* p = W + (size_t)(n0 + (lane & 15)) * ldw + k0 + ((lane & 16) ? 8 : 0);
  v16bf r;
#pragma unroll
  for (int i = 0; i < 8; ++i) r[i] = (__bf16)p[i];
#pragma unroll
  for (int i = 0; i < 8; ++i) r[8 + i] = (__bf16)p[16 + i];
  return r;
}

__device__ __forceinline__ float fast_silu(float x) {
  return x / (1.0f + __expf(-x));
}

// ---------------- K0: pad x_proj_weight (4,38,192) -> (4,48,192) -----------
__global__ void k_padw(const float* __restrict__ xpw, float* __restrict__ wpad) {
  int idx = blockIdx.x * blockDim.x + threadIdx.x;   // over 4*48*192
  if (idx >= KDIR * XLD * DI) return;
  int j = idx % DI;
  int c = (idx / DI) % XLD;
  int k = idx / (DI * XLD);
  wpad[idx] = (c < CDBL) ? xpw[((size_t)k * CDBL + c) * DI + j] : 0.0f;
}

// ---------------- K1: in_proj GEMM  xz = x @ in_proj_w^T -------------------
// x: (B*L, 96) row-major. W: (384, 96). Out: xq planar (b,d,h,w) + z (b,l,192).
// Each wave: one 16-row M tile x 4 consecutive N tiles (64 cols).
__global__ void k_inproj(const float* __restrict__ x, const float* __restrict__ w,
                         float* __restrict__ xq, float* __restrict__ z) {
  const int NG = 6;                             // 24 n-tiles / 4 per wave
  int wave = threadIdx.x >> 5, lane = threadIdx.x & 31;
  int job = blockIdx.x * 4 + wave;              // 1024 * 6 jobs
  int mt = job / NG, ng = job % NG;
  int m0 = mt * 16, n0base = ng * 64;
  v8f acc[4] = {};
#pragma unroll
  for (int k0 = 0; k0 < DMODEL; k0 += 32) {
    v16bf at = load_a_rm(x, DMODEL, m0, k0, lane);
#pragma unroll
    for (int j = 0; j < 4; ++j) {
      v16bf bt = load_bT_rm(w, DMODEL, n0base + j * 16, k0, lane);
      acc[j] = wmma_bf16(at, bt, acc[j]);
    }
  }
  int nl = lane & 15;
  int mbase = m0 + ((lane >> 4) << 3);
#pragma unroll
  for (int j = 0; j < 4; ++j) {
    int n = n0base + j * 16 + nl;
#pragma unroll
    for (int r = 0; r < 8; ++r) {
      int m = mbase + r;                 // m = b*4096 + l
      int bb = m >> 12, l = m & (LSEQ - 1);
      float v = acc[j][r];
      if (n < DI) xq[((size_t)(bb * DI + n)) * LSEQ + l] = v;
      else        z[(size_t)m * DI + (n - DI)] = v;
    }
  }
}

// ---------------- K2: depthwise 3x3 conv + SiLU + 4-direction scatter ------
__global__ void k_conv(const float* __restrict__ xq, const float* __restrict__ cw,
                       const float* __restrict__ cb, float* __restrict__ xs) {
  int idx = blockIdx.x * blockDim.x + threadIdx.x;   // over B*DI*L
  if (idx >= BQ * DI * LSEQ) return;
  int l = idx & (LSEQ - 1);
  int d = (idx >> 12) % DI;
  int b = idx / (DI * LSEQ);
  int h = l >> 6, ww = l & 63;
  const float* plane = xq + (size_t)(b * DI + d) * LSEQ;
  const float* k9 = cw + d * 9;
  float acc = cb[d];
#pragma unroll
  for (int dy = -1; dy <= 1; ++dy)
#pragma unroll
    for (int dx = -1; dx <= 1; ++dx) {
      int hh = h + dy, wx = ww + dx;
      if (hh >= 0 && hh < HH && wx >= 0 && wx < WWID)
        acc += k9[(dy + 1) * 3 + (dx + 1)] * plane[hh * WWID + wx];
    }
  float v = fast_silu(acc);
  int lhw = h * WWID + ww, lwh = ww * HH + h;
  size_t base = (size_t)b * (KDIR * LSEQ * DI);
  xs[base + ((size_t)(0 * LSEQ + lhw)) * DI + d] = v;
  xs[base + ((size_t)(1 * LSEQ + lwh)) * DI + d] = v;
  xs[base + ((size_t)(2 * LSEQ + (LSEQ - 1 - lhw))) * DI + d] = v;
  xs[base + ((size_t)(3 * LSEQ + (LSEQ - 1 - lwh))) * DI + d] = v;
}

// ---------------- K3: x_dbl GEMM  (per b,k): xs(L,192) @ wpad[k]^T (48,192) -
// Each wave: one 16-row M tile x all 3 N tiles.
__global__ void k_xdbl(const float* __restrict__ xs, const float* __restrict__ wpad,
                       float* __restrict__ xdbl) {
  const int MT = LSEQ / 16;          // 256
  int wave = threadIdx.x >> 5, lane = threadIdx.x & 31;
  int job = blockIdx.x * 4 + wave;   // 16 * 256 jobs
  int bk = job / MT;
  int mt = job % MT;
  const float* A = xs + (size_t)bk * LSEQ * DI;
  const float* W = wpad + (size_t)(bk & 3) * XLD * DI;
  int m0 = mt * 16;
  v8f acc[3] = {};
#pragma unroll
  for (int k0 = 0; k0 < DI; k0 += 32) {
    v16bf at = load_a_rm(A, DI, m0, k0, lane);
#pragma unroll
    for (int j = 0; j < 3; ++j) {
      v16bf bt = load_bT_rm(W, DI, j * 16, k0, lane);
      acc[j] = wmma_bf16(at, bt, acc[j]);
    }
  }
  int nl = lane & 15;
  int mbase = m0 + ((lane >> 4) << 3);
#pragma unroll
  for (int j = 0; j < 3; ++j) {
    int n = j * 16 + nl;
    if (n < CDBL) {
#pragma unroll
      for (int r = 0; r < 8; ++r)
        xdbl[((size_t)bk * LSEQ + mbase + r) * XLD + n] = acc[j][r];
    }
  }
}

// ---------------- Selective scan (chunked, 3-pass) -------------------------
// h_l[n] = exp(delta_l*A[n]) * h_{l-1}[n] + delta_l*u_l * Bs_l[n]
// y_l    = sum_n h_l[n]*Cs_l[n] + u_l*Ds

// Pass 1: per (b,k,chunk): P[n] = prod dA, S[n] = h after chunk from h=0.
__global__ void k_scan_pass1(const float* __restrict__ xs, const float* __restrict__ xdbl,
                             const float* __restrict__ dtw, const float* __restrict__ dtb,
                             const float* __restrict__ Alogs,
                             float* __restrict__ chP, float* __restrict__ chS) {
  int d = threadIdx.x;
  int c = blockIdx.x % NC;
  int bk = blockIdx.x / NC;
  int kk = bk & 3;
  float Ar[NST];
  const float* al = Alogs + (size_t)(kk * DI + d) * NST;
#pragma unroll
  for (int n = 0; n < NST; ++n) Ar[n] = -__expf(al[n]);
  float w6[DTR];
  const float* pw = dtw + (size_t)(kk * DI + d) * DTR;
#pragma unroll
  for (int r = 0; r < DTR; ++r) w6[r] = pw[r];
  float bias = dtb[kk * DI + d];
  float h[NST], P[NST];
#pragma unroll
  for (int n = 0; n < NST; ++n) { h[n] = 0.f; P[n] = 1.f; }
  size_t rowbase = (size_t)bk * LSEQ + (size_t)c * CLEN;
  const float* urow = xs + rowbase * DI + d;
  const float* xrow = xdbl + rowbase * XLD;
  for (int t = 0; t < CLEN; ++t) {
    const float* row = xrow + (size_t)t * XLD;
    if (t + 1 < CLEN) __builtin_prefetch(row + XLD, 0, 0);   // global_prefetch_b8
    float dtv = bias;
#pragma unroll
    for (int r = 0; r < DTR; ++r) dtv += row[r] * w6[r];
    float delta = (dtv > 20.f) ? dtv : __logf(1.f + __expf(dtv));  // softplus
    float du = delta * urow[(size_t)t * DI];
#pragma unroll
    for (int n = 0; n < NST; ++n) {
      float dA = __expf(delta * Ar[n]);
      P[n] *= dA;
      h[n] = h[n] * dA + du * row[DTR + n];
    }
  }
  size_t obase = ((size_t)blockIdx.x * DI + d) * NST;   // (bk, c, d, n)
#pragma unroll
  for (int n = 0; n < NST; ++n) { chP[obase + n] = P[n]; chS[obase + n] = h[n]; }
}

// Pass 2: sequential combine over chunks; Hstart[bk,c,d,n] = h at chunk start.
__global__ void k_scan_pass2(const float* __restrict__ chP, const float* __restrict__ chS,
                             float* __restrict__ Hstart) {
  int idx = blockIdx.x * blockDim.x + threadIdx.x;   // over 16*192*16
  if (idx >= BQ * KDIR * DI * NST) return;
  int bk = idx / (DI * NST);
  int dn = idx % (DI * NST);
  float hp = 0.f;
  for (int c = 0; c < NC; ++c) {
    size_t off = ((size_t)(bk * NC + c)) * (DI * NST) + dn;
    Hstart[off] = hp;
    hp = chP[off] * hp + chS[off];
  }
}

// Pass 3: replay each chunk from Hstart, emit y (in place over xs).
__global__ void k_scan_pass3(float* __restrict__ xs_y, const float* __restrict__ xdbl,
                             const float* __restrict__ dtw, const float* __restrict__ dtb,
                             const float* __restrict__ Alogs, const float* __restrict__ Ds,
                             const float* __restrict__ Hstart) {
  int d = threadIdx.x;
  int c = blockIdx.x % NC;
  int bk = blockIdx.x / NC;
  int kk = bk & 3;
  float Ar[NST];
  const float* al = Alogs + (size_t)(kk * DI + d) * NST;
#pragma unroll
  for (int n = 0; n < NST; ++n) Ar[n] = -__expf(al[n]);
  float w6[DTR];
  const float* pw = dtw + (size_t)(kk * DI + d) * DTR;
#pragma unroll
  for (int r = 0; r < DTR; ++r) w6[r] = pw[r];
  float bias = dtb[kk * DI + d];
  float Dsv = Ds[kk * DI + d];
  float h[NST];
  size_t hoff = ((size_t)blockIdx.x * DI + d) * NST;
#pragma unroll
  for (int n = 0; n < NST; ++n) h[n] = Hstart[hoff + n];
  size_t rowbase = (size_t)bk * LSEQ + (size_t)c * CLEN;
  float* urow = xs_y + rowbase * DI + d;
  const float* xrow = xdbl + rowbase * XLD;
  for (int t = 0; t < CLEN; ++t) {
    const float* row = xrow + (size_t)t * XLD;
    if (t + 1 < CLEN) __builtin_prefetch(row + XLD, 0, 0);
    float dtv = bias;
#pragma unroll
    for (int r = 0; r < DTR; ++r) dtv += row[r] * w6[r];
    float delta = (dtv > 20.f) ? dtv : __logf(1.f + __expf(dtv));
    float u = urow[(size_t)t * DI];
    float du = delta * u;
    float y = 0.f;
#pragma unroll
    for (int n = 0; n < NST; ++n) {
      float dA = __expf(delta * Ar[n]);
      h[n] = h[n] * dA + du * row[DTR + n];
      y += h[n] * row[DTR + NST + n];
    }
    urow[(size_t)t * DI] = y + u * Dsv;   // overwrite xs with y (same element)
  }
}

// ---------------- K7: direction combine + LayerNorm + SiLU gate ------------
__global__ void k_combine_ln(const float* __restrict__ y, const float* __restrict__ z,
                             const float* __restrict__ g, const float* __restrict__ beta,
                             float* __restrict__ yln) {
  int bl = blockIdx.x;                  // b*4096 + l
  int b = bl >> 12, l = bl & (LSEQ - 1);
  int t = threadIdx.x;
  int h = l >> 6, ww = l & 63;
  int lwh = ww * HH + h;
  float v = 0.f;
  if (t < DI) {
    size_t base = (size_t)b * KDIR * LSEQ * DI;
    v = y[base + ((size_t)(0 * LSEQ + l)) * DI + t]
      + y[base + ((size_t)(2 * LSEQ + (LSEQ - 1 - l))) * DI + t]
      + y[base + ((size_t)(1 * LSEQ + lwh)) * DI + t]
      + y[base + ((size_t)(3 * LSEQ + (LSEQ - 1 - lwh))) * DI + t];
  }
  __shared__ float red[256];
  red[t] = (t < DI) ? v : 0.f;
  __syncthreads();
  for (int s = 128; s > 0; s >>= 1) { if (t < s) red[t] += red[t + s]; __syncthreads(); }
  float mu = red[0] * (1.0f / DI);
  __syncthreads();
  float dv = (t < DI) ? (v - mu) : 0.f;
  red[t] = dv * dv;
  __syncthreads();
  for (int s = 128; s > 0; s >>= 1) { if (t < s) red[t] += red[t + s]; __syncthreads(); }
  float var = red[0] * (1.0f / DI);
  float rs = rsqrtf(var + 1e-5f);
  if (t < DI) {
    float zz = z[(size_t)bl * DI + t];
    yln[(size_t)bl * DI + t] = ((v - mu) * rs * g[t] + beta[t]) * fast_silu(zz);
  }
}

// ---------------- K8: out_proj GEMM  out = yln @ out_proj_w^T --------------
// Each wave: one 16-row M tile x all 6 N tiles (full 96-col output row block).
__global__ void k_outproj(const float* __restrict__ yln, const float* __restrict__ w,
                          float* __restrict__ out) {
  int wave = threadIdx.x >> 5, lane = threadIdx.x & 31;
  int mt = blockIdx.x * 4 + wave;       // 1024 m-tiles
  int m0 = mt * 16;
  v8f acc[6] = {};
#pragma unroll
  for (int k0 = 0; k0 < DI; k0 += 32) {
    v16bf at = load_a_rm(yln, DI, m0, k0, lane);
#pragma unroll
    for (int j = 0; j < 6; ++j) {
      v16bf bt = load_bT_rm(w, DI, j * 16, k0, lane);
      acc[j] = wmma_bf16(at, bt, acc[j]);
    }
  }
  int nl = lane & 15;
  int mbase = m0 + ((lane >> 4) << 3);
#pragma unroll
  for (int j = 0; j < 6; ++j) {
    int n = j * 16 + nl;
#pragma unroll
    for (int r = 0; r < 8; ++r)
      out[(size_t)(mbase + r) * DMODEL + n] = acc[j][r];
  }
}

// ---------------------------------------------------------------------------

extern "C" void kernel_launch(void* const* d_in, const int* in_sizes, int n_in,
                              void* d_out, int out_size, void* d_ws, size_t ws_size,
                              hipStream_t stream) {
  (void)in_sizes; (void)n_in; (void)out_size; (void)ws_size;
  const float* x    = (const float*)d_in[0];   // (B,64,64,96)
  const float* ipw  = (const float*)d_in[1];   // (384,96)
  const float* cw   = (const float*)d_in[2];   // (192,1,3,3)
  const float* cb   = (const float*)d_in[3];   // (192,)
  const float* xpw  = (const float*)d_in[4];   // (4,38,192)
  const float* dtw  = (const float*)d_in[5];   // (4,192,6)
  const float* dtb  = (const float*)d_in[6];   // (4,192)
  const float* alog = (const float*)d_in[7];   // (768,16)
  const float* Ds   = (const float*)d_in[8];   // (768,)
  const float* og   = (const float*)d_in[9];   // (192,)
  const float* ob   = (const float*)d_in[10];  // (192,)
  const float* opw  = (const float*)d_in[11];  // (96,192)
  float* out = (float*)d_out;

  // Workspace layout (floats); total ~26.8M floats (~107 MB).
  float* ws   = (float*)d_ws;
  float* wq   = ws;                         // xq planar (B,DI,L); reused as yln
  float* wz   = wq   + (size_t)BQ * DI * LSEQ;          // z (B,L,DI)
  float* wxs  = wz   + (size_t)BQ * LSEQ * DI;          // xs (B,K,L,DI); reused as y
  float* wdbl = wxs  + (size_t)BQ * KDIR * LSEQ * DI;   // x_dbl (B*K, L, 48)
  float* wP   = wdbl + (size_t)BQ * KDIR * LSEQ * XLD;  // chunk products
  float* wS   = wP   + (size_t)BQ * KDIR * NC * DI * NST;
  float* wH   = wS   + (size_t)BQ * KDIR * NC * DI * NST;
  float* wpad = wH   + (size_t)BQ * KDIR * NC * DI * NST;  // (4,48,192) padded xpw
  float* yln  = wq;                         // alias: xq dead after conv

  // K0: pad x_proj_weight so x_dbl B-loads need no bounds guard.
  k_padw<<<(KDIR * XLD * DI + 255) / 256, 256, 0, stream>>>(xpw, wpad);
  // K1: in_proj. 1024 m-tiles * 6 n-groups (4 tiles each), 4 waves/block.
  k_inproj<<<(1024 * 6) / 4, 128, 0, stream>>>(x, ipw, wq, wz);
  // K2: depthwise conv + SiLU + scatter into 4 scan orders.
  k_conv<<<(BQ * DI * LSEQ) / 256, 256, 0, stream>>>(wq, cw, cb, wxs);
  // K3: x_dbl per-direction GEMM. 16 bk * 256 m-tiles, 3 n-tiles/wave.
  k_xdbl<<<(16 * 256) / 4, 128, 0, stream>>>(wxs, wpad, wdbl);
  // K4-K6: chunked selective scan.
  k_scan_pass1<<<BQ * KDIR * NC, DI, 0, stream>>>(wxs, wdbl, dtw, dtb, alog, wP, wS);
  k_scan_pass2<<<(BQ * KDIR * DI * NST) / 256, 256, 0, stream>>>(wP, wS, wH);
  k_scan_pass3<<<BQ * KDIR * NC, DI, 0, stream>>>(wxs, wdbl, dtw, dtb, alog, Ds, wH);
  // K7: combine 4 directions + LayerNorm + SiLU(z) gate.
  k_combine_ln<<<BQ * LSEQ, 256, 0, stream>>>(wxs, wz, og, ob, yln);
  // K8: out_proj to final output (B,64,64,96).
  k_outproj<<<(1024 * 6) / 4 / 6, 128, 0, stream>>>(yln, opw, out);
}
